// GQA_19825569038644
// MI455X (gfx1250) — matrix operand order repaired and checked by reading, not compiled, not run on hardware
//
#include <hip/hip_runtime.h>

typedef __attribute__((ext_vector_type(16))) __bf16 v16bf;
typedef __attribute__((ext_vector_type(8)))  __bf16 v8bf;
typedef __attribute__((ext_vector_type(2)))  __bf16 v2bf;
typedef __attribute__((ext_vector_type(8)))  float  v8f;
typedef __attribute__((ext_vector_type(4)))  int    v4i;

#if defined(__has_builtin)
#  if __has_builtin(__builtin_amdgcn_global_load_async_to_lds_b128)
#    define HAVE_ASYNC_LDS 1
#  endif
#  if __has_builtin(__builtin_amdgcn_cvt_pk_bf16_f32)
#    define HAVE_PK_BF16 1
#  endif
#endif
#ifndef HAVE_ASYNC_LDS
#  define HAVE_ASYNC_LDS 0
#endif
#ifndef HAVE_PK_BF16
#  define HAVE_PK_BF16 0
#endif

typedef __attribute__((address_space(1))) v4i as1_v4i;   // global int4 (clang: "int4 __device__*")
typedef __attribute__((address_space(3))) v4i as3_v4i;   // LDS int4

__device__ __forceinline__ __bf16 f2bf(float f) { return (__bf16)f; }  // native cvt

__device__ __forceinline__ v2bf pk_bf16(float a, float b) {
#if HAVE_PK_BF16
  return __builtin_amdgcn_cvt_pk_bf16_f32(a, b);
#else
  v2bf r; r[0] = (__bf16)a; r[1] = (__bf16)b; return r;
#endif
}

__device__ __forceinline__ v16bf cat8(v8bf lo, v8bf hi) {
  return __builtin_shufflevector(lo, hi, 0,1,2,3,4,5,6,7,8,9,10,11,12,13,14,15);
}

// ---------------------------------------------------------------------------
// LDS-staged block GEMM  Out = A[M,K] * Bw[K,N]  (bf16 WMMA, f32 accumulate)
// Block: 256 threads = 8 waves, tile 128(M) x 128(N), K-step 32.
//  - A staged row-major into padded LDS (row stride 40 bf16 -> conflict-free
//    16B fragment reads).  bf16 A uses GLOBAL_LOAD_ASYNC_TO_LDS_B128 when the
//    toolchain exposes it (ASYNCcnt path), else vector loads.
//  - B staged TRANSPOSED so the WMMA B-fragment is 2 contiguous ds_load_b128.
// Wave tile: 64(M) x 32(N) = 4x2 accumulators, 8 wmma per K-step.
// ---------------------------------------------------------------------------
enum { OUT_F32 = 0, OUT_BF16 = 1, OUT_BF16_T = 2 };
constexpr int BM = 128, BN = 128, BK = 32, LDP = 40;  // LDP: padded LDS row (bf16)

template<bool A_BF16, int OMODE>
__global__ __launch_bounds__(256, 1)
void gqa_gemm(const void* __restrict__ Ap, const float* __restrict__ Bw,
              void* __restrict__ Outp, int M, int N, int K, int ldo) {
  __shared__ __bf16 As[BM * LDP];
  __shared__ __bf16 BsT[BN * LDP];

  const int tid  = threadIdx.x;
  const int lane = tid & 31;
  const int wid  = tid >> 5;          // 0..7
  const int hlf  = lane >> 4;
  const int l16  = lane & 15;
  const int wm   = wid & 1;           // 2 waves along M
  const int wn   = wid >> 1;          // 4 waves along N
  const int mB   = blockIdx.y * BM;
  const int nB   = blockIdx.x * BN;
  const int mWo  = wm * 64;
  const int nWo  = wn * 32;

  const float*  Af = (const float*)Ap;
  const __bf16* Ab = (const __bf16*)Ap;

  v8f acc[4][2] = {};

  const int arow = tid >> 1, ach = (tid & 1) * 16;   // A: 128 rows x 32 cols
  const int brow = tid >> 3, bch = (tid & 7) * 16;   // B: 32 k-rows x 128 cols

  for (int k0 = 0; k0 < K; k0 += BK) {
    __syncthreads();   // previous iteration's fragment reads complete
    // ---- stage A tile ----
    {
      __bf16* dst = &As[arow * LDP + ach];
      if constexpr (A_BF16) {
        const __bf16* src = Ab + (size_t)(mB + arow) * K + k0 + ach;
#if HAVE_ASYNC_LDS
        as1_v4i* gsrc = (as1_v4i*)(v4i*)(void*)const_cast<__bf16*>(src);
        as3_v4i* ldst = (as3_v4i*)(v4i*)(void*)dst;
        __builtin_amdgcn_global_load_async_to_lds_b128(gsrc, ldst, 0, 0);
        __builtin_amdgcn_global_load_async_to_lds_b128(gsrc, ldst, 16, 0);
#else
        *(v8bf*)dst       = *(const v8bf*)src;
        *(v8bf*)(dst + 8) = *(const v8bf*)(src + 8);
#endif
      } else {
        const float* src = Af + (size_t)(mB + arow) * K + k0 + ach;
        union { v2bf p[4]; v8bf v; } lo, hi;
#pragma unroll
        for (int i = 0; i < 4; ++i) {
          lo.p[i] = pk_bf16(src[2 * i],     src[2 * i + 1]);
          hi.p[i] = pk_bf16(src[8 + 2 * i], src[9 + 2 * i]);
        }
        *(v8bf*)dst       = lo.v;
        *(v8bf*)(dst + 8) = hi.v;
      }
    }
    // ---- stage B tile transposed ----
    {
      const float* src = Bw + (size_t)(k0 + brow) * N + nB + bch;
#pragma unroll
      for (int i = 0; i < 16; ++i) BsT[(bch + i) * LDP + brow] = f2bf(src[i]);
    }
    // ---- prefetch next K tile (global_prefetch_b8) ----
    if (k0 + BK < K) {
      if constexpr (A_BF16) __builtin_prefetch(Ab + (size_t)(mB + arow) * K + k0 + BK + ach, 0, 1);
      else                  __builtin_prefetch(Af + (size_t)(mB + arow) * K + k0 + BK + ach, 0, 1);
      __builtin_prefetch(Bw + (size_t)(k0 + BK + brow) * N + nB + bch, 0, 1);
    }
#if HAVE_ASYNC_LDS
    if constexpr (A_BF16) asm volatile("s_wait_asynccnt 0" ::: "memory");
#endif
    __syncthreads();

    // ---- B fragments: lane = column, contiguous K in transposed LDS ----
    v16bf bfrag[2];
#pragma unroll
    for (int ni = 0; ni < 2; ++ni) {
      const __bf16* p = &BsT[(nWo + ni * 16 + l16) * LDP + hlf * 8];
      bfrag[ni] = cat8(*(const v8bf*)p, *(const v8bf*)(p + 16));
    }
    // ---- A fragments + WMMA ----
#pragma unroll
    for (int mi = 0; mi < 4; ++mi) {
      const __bf16* p = &As[(mWo + mi * 16 + l16) * LDP + hlf * 8];
      v16bf afrag = cat8(*(const v8bf*)p, *(const v8bf*)(p + 16));
#pragma unroll
      for (int ni = 0; ni < 2; ++ni)
        acc[mi][ni] = __builtin_amdgcn_wmma_f32_16x16x32_bf16(
            false, afrag, false, bfrag[ni], (short)0, acc[mi][ni], false, false);
    }
  }

  // ---- store (C layout: VGPR j -> row j + hlf*8, col = l16) ----
#pragma unroll
  for (int mi = 0; mi < 4; ++mi)
#pragma unroll
    for (int ni = 0; ni < 2; ++ni) {
      const int col  = nB + nWo + ni * 16 + l16;
      const int row0 = mB + mWo + mi * 16 + hlf * 8;
      if constexpr (OMODE == OUT_BF16_T) {
        union { v2bf p[4]; v8bf v; } ov;
#pragma unroll
        for (int j = 0; j < 4; ++j)
          ov.p[j] = pk_bf16(acc[mi][ni][2 * j], acc[mi][ni][2 * j + 1]);
        *(v8bf*)((__bf16*)Outp + (size_t)col * ldo + row0) = ov.v;   // contiguous 16B
      } else if constexpr (OMODE == OUT_BF16) {
#pragma unroll
        for (int j = 0; j < 8; ++j)
          ((__bf16*)Outp)[(size_t)(row0 + j) * ldo + col] = f2bf(acc[mi][ni][j]);
      } else {
#pragma unroll
        for (int j = 0; j < 8; ++j)
          ((float*)Outp)[(size_t)(row0 + j) * ldo + col] = acc[mi][ni][j];
      }
    }
}

// ---------------------------------------------------------------------------
// Flash attention (causal, GQA). One wave = one 16-query tile of one (b,head).
// q: [B*T, 2048] bf16 (head h at col h*64); k: [B*T, 512] bf16 (kv head g*64)
// vT: [512, B*T] bf16 (transposed) -> P*V B-fragments are contiguous loads.
// Softmax runs in the exp2 domain: scale2 = (1/sqrt(64)) * log2(e).
// ---------------------------------------------------------------------------
__global__ __launch_bounds__(128, 1)
void gqa_flash_attn(const __bf16* __restrict__ qb,
                    const __bf16* __restrict__ kbuf,
                    const __bf16* __restrict__ vT,
                    __bf16* __restrict__ ob) {
  constexpr int T = 2048, DQ = 2048, DKV = 512, HD = 64, MT = 4096;  // MT = B*T
  __shared__ __bf16 plds[4][16 * 32];            // per-wave P staging

  const int lane = threadIdx.x & 31;
  const int hlf  = lane >> 4;
  const int l16  = lane & 15;
  const int wv   = threadIdx.y;
  const int q0   = (blockIdx.x * 4 + wv) * 16;
  const int h    = blockIdx.y;
  const int b    = blockIdx.z;
  const int g    = h >> 2;                       // kv head (REP = 4)

  // ---- Q as two 16x32 A-tiles (dims 0..31, 32..63) ----
  const __bf16* qrow = qb + (size_t)(b * T + q0 + l16) * DQ + h * HD;
  v16bf qa[2];
#pragma unroll
  for (int s = 0; s < 2; ++s)
    qa[s] = cat8(*(const v8bf*)(qrow + s * 32 + hlf * 8),
                 *(const v8bf*)(qrow + s * 32 + 16 + hlf * 8));

  float mrow[8], lrow[8];
#pragma unroll
  for (int j = 0; j < 8; ++j) { mrow[j] = -1e30f; lrow[j] = 0.f; }
  v8f o[4] = {};
  const float scale2 = 0.125f * 1.44269504088896341f;   // 1/sqrt(hd) * log2(e)

  for (int kb = 0; kb < q0 + 16; kb += 32) {
    // ---- S = Q K^T : two 16x16 f32 tiles over 32 keys ----
    v8f sacc[2] = {};
#pragma unroll
    for (int tt = 0; tt < 2; ++tt) {
      const __bf16* krow = kbuf + (size_t)(b * T + kb + tt * 16 + l16) * DKV + g * HD;
#pragma unroll
      for (int s = 0; s < 2; ++s) {
        v16bf bt = cat8(*(const v8bf*)(krow + s * 32 + hlf * 8),
                        *(const v8bf*)(krow + s * 32 + 16 + hlf * 8));
        sacc[tt] = __builtin_amdgcn_wmma_f32_16x16x32_bf16(
            false, qa[s], false, bt, (short)0, sacc[tt], false, false);
      }
    }

    // ---- online softmax, exp2 domain (16-lane shuffle reductions per row) ----
#pragma unroll
    for (int j = 0; j < 8; ++j) {
      const int row = q0 + hlf * 8 + j;
      float s0 = sacc[0][j] * scale2;
      float s1 = sacc[1][j] * scale2;
      if (kb + l16 > row)      s0 = -1e30f;      // causal mask
      if (kb + 16 + l16 > row) s1 = -1e30f;
      float mx = fmaxf(s0, s1);
#pragma unroll
      for (int off = 8; off; off >>= 1) mx = fmaxf(mx, __shfl_xor(mx, off, 16));
      const float mnew  = fmaxf(mrow[j], mx);
      const float alpha = exp2f(mrow[j] - mnew); // bare v_exp_f32
      const float p0 = exp2f(s0 - mnew);
      const float p1 = exp2f(s1 - mnew);
      float rs = p0 + p1;
#pragma unroll
      for (int off = 8; off; off >>= 1) rs += __shfl_xor(rs, off, 16);
      mrow[j] = mnew;
      lrow[j] = lrow[j] * alpha + rs;
#pragma unroll
      for (int t = 0; t < 4; ++t) o[t][j] *= alpha;
      __bf16* pr = &plds[wv][(hlf * 8 + j) * 32];
      pr[l16]      = f2bf(p0);
      pr[16 + l16] = f2bf(p1);
    }
    asm volatile("s_wait_dscnt 0" ::: "memory");   // CDNA5 split LDS counter fence

    // ---- P back as one 16x32 A-tile ----
    const __bf16* pr = &plds[wv][l16 * 32 + hlf * 8];
    v16bf pa = cat8(*(const v8bf*)pr, *(const v8bf*)(pr + 16));

    // ---- O += P V : V^T layout makes B-fragments contiguous ----
#pragma unroll
    for (int t = 0; t < 4; ++t) {
      const __bf16* vp = vT + (size_t)(g * HD + t * 16 + l16) * MT + b * T + kb + hlf * 8;
      v16bf vt = cat8(*(const v8bf*)vp, *(const v8bf*)(vp + 16));
      o[t] = __builtin_amdgcn_wmma_f32_16x16x32_bf16(
          false, pa, false, vt, (short)0, o[t], false, false);
    }
  }

  // ---- normalize + store bf16 head-major ----
#pragma unroll
  for (int j = 0; j < 8; ++j) {
    const float inv = __builtin_amdgcn_rcpf(lrow[j]);
    const size_t row = (size_t)(b * T + q0 + hlf * 8 + j);
#pragma unroll
    for (int t = 0; t < 4; ++t)
      ob[row * DQ + h * HD + t * 16 + l16] = f2bf(o[t][j] * inv);
  }
}

// ---------------------------------------------------------------------------
extern "C" void kernel_launch(void* const* d_in, const int* in_sizes, int n_in,
                              void* d_out, int out_size, void* d_ws, size_t ws_size,
                              hipStream_t stream) {
  (void)in_sizes; (void)n_in; (void)out_size; (void)ws_size;
  const float* x  = (const float*)d_in[0];
  const float* Wq = (const float*)d_in[1];
  const float* Wk = (const float*)d_in[2];
  const float* Wv = (const float*)d_in[3];
  const float* Wo = (const float*)d_in[4];
  float* out = (float*)d_out;

  constexpr int B = 2, T = 2048, D = 2048, H = 32, G = 8, HD = 64;
  constexpr int M   = B * T;       // 4096
  constexpr int NKV = G * HD;      // 512

  char* ws = (char*)d_ws;
  __bf16* qbuf = (__bf16*)(ws);                                                 // [M, D]
  __bf16* kbuf = (__bf16*)(ws + (size_t)M * D * 2);                             // [M, NKV]
  __bf16* vTbf = (__bf16*)(ws + (size_t)M * D * 2 + (size_t)M * NKV * 2);       // [NKV, M]
  __bf16* abuf = (__bf16*)(ws + (size_t)M * D * 2 + 2 * (size_t)M * NKV * 2);   // [M, D]

  dim3 gblk(256);

  // QKV projections (fp32 x, fp32 W -> bf16 activations); V written transposed
  gqa_gemm<false, OUT_BF16  ><<<dim3(D   / BN, M / BM), gblk, 0, stream>>>(x, Wq, qbuf, M, D,   D, D);
  gqa_gemm<false, OUT_BF16  ><<<dim3(NKV / BN, M / BM), gblk, 0, stream>>>(x, Wk, kbuf, M, NKV, D, NKV);
  gqa_gemm<false, OUT_BF16_T><<<dim3(NKV / BN, M / BM), gblk, 0, stream>>>(x, Wv, vTbf, M, NKV, D, M);

  // causal GQA flash attention
  gqa_flash_attn<<<dim3(T / 64, H, B), dim3(32, 4), 0, stream>>>(qbuf, kbuf, vTbf, abuf);

  // output projection: bf16 attn @ fp32 Wo -> fp32 out
  gqa_gemm<true, OUT_F32><<<dim3(D / BN, M / BM), gblk, 0, stream>>>(abuf, Wo, out, M, D, D, D);
}
